// MHA_12592844112296
// MI455X (gfx1250) — matrix operand order repaired and checked by reading, not compiled, not run on hardware
//
#include <hip/hip_runtime.h>
#include <hip/hip_bf16.h>
#include <stdint.h>

// MHA forward for MI455X (gfx1250). All matmuls on V_WMMA_F32_16X16X32_BF16;
// GEMM LDS staging uses gfx1250 async global->LDS copies when available.
#define B_  32
#define T_  576
#define C_  1024
#define H_  16
#define D_  64
#define N3_ 3072

typedef __attribute__((ext_vector_type(16))) __bf16 v16bf;
typedef __attribute__((ext_vector_type(8)))  float  v8f;

union Frag16 { v16bf bf; uint4 u[2]; };

__device__ __forceinline__ unsigned short f2bf(float f) {
  unsigned int u = __float_as_uint(f);
  u += 0x7fffu + ((u >> 16) & 1u);          // round-to-nearest-even
  return (unsigned short)(u >> 16);
}

__device__ __forceinline__ v8f wmma_bf16(const Frag16& a, const Frag16& b, v8f c) {
  return __builtin_amdgcn_wmma_f32_16x16x32_bf16(
      false, a.bf, false, b.bf, (short)0, c, false, false);
}

// ---- gfx1250 async global->LDS (ASYNCcnt) with sync fallback -------------
#if __has_builtin(__builtin_amdgcn_global_load_async_to_lds_b128)
#define USE_ASYNC_LDS 1
// Builtin signature (from hipcc diagnostic): param0 = AS1 (global) pointer to
// int __attribute__((vector_size(16))), param1 = AS3 (LDS) same pointee.
typedef int v4i_gnu __attribute__((vector_size(16)));
typedef __attribute__((address_space(1))) v4i_gnu as1_v4i;
typedef __attribute__((address_space(3))) v4i_gnu as3_v4i;
__device__ __forceinline__ void cp16(const unsigned short* g, unsigned short* l) {
  // LDS generic pointer: low 32 bits are the LDS byte offset (ISA aperture rule).
  __builtin_amdgcn_global_load_async_to_lds_b128(
      (as1_v4i*)(unsigned long long)(uintptr_t)g,
      (as3_v4i*)(unsigned int)(uintptr_t)l, 0, 0);
}
__device__ __forceinline__ void wait_async_hi() {
#if __has_builtin(__builtin_amdgcn_s_wait_asynccnt)
  __builtin_amdgcn_s_wait_asynccnt(6);
#else
  asm volatile("s_wait_asynccnt 6" ::: "memory");
#endif
}
__device__ __forceinline__ void wait_async_0() {
#if __has_builtin(__builtin_amdgcn_s_wait_asynccnt)
  __builtin_amdgcn_s_wait_asynccnt(0);
#else
  asm volatile("s_wait_asynccnt 0" ::: "memory");
#endif
}
#else
__device__ __forceinline__ void cp16(const unsigned short* g, unsigned short* l) {
  *(uint4*)l = *(const uint4*)g;
}
__device__ __forceinline__ void wait_async_hi() {}
__device__ __forceinline__ void wait_async_0() {}
#endif

// ---------------------------------------------------------------------------
// One-time converts: f32 -> bf16 (X), and f32 -> bf16 transposed (weights).
// ---------------------------------------------------------------------------
__global__ __launch_bounds__(256) void convert_bf16_kernel(
    const float* __restrict__ src, unsigned short* __restrict__ dst, int n8) {
  const int i = blockIdx.x * 256 + threadIdx.x;
  if (i >= n8) return;
  const size_t off = (size_t)i * 8;
  float4 a0 = *(const float4*)(src + off);
  float4 a1 = *(const float4*)(src + off + 4);
  union { unsigned short s[8]; uint4 u; } pk;
  pk.s[0] = f2bf(a0.x); pk.s[1] = f2bf(a0.y); pk.s[2] = f2bf(a0.z); pk.s[3] = f2bf(a0.w);
  pk.s[4] = f2bf(a1.x); pk.s[5] = f2bf(a1.y); pk.s[6] = f2bf(a1.z); pk.s[7] = f2bf(a1.w);
  *(uint4*)(dst + off) = pk.u;
}

// W [C_][N] f32 -> Wt [N][C_] bf16, tiled through LDS.
__global__ __launch_bounds__(256) void convert_transpose_kernel(
    const float* __restrict__ W, unsigned short* __restrict__ Wt, int N) {
  __shared__ unsigned short Ws[64][72];
  const int tid = threadIdx.x;
  const int nbase = blockIdx.x * 64;
  const int kbase = blockIdx.y * 64;
  const int row = tid >> 2;                 // k-local 0..63
  const int c16 = (tid & 3) * 16;          // n-local
  const float* wp = &W[(size_t)(kbase + row) * N + nbase + c16];
  #pragma unroll
  for (int j = 0; j < 2; ++j) {
    float4 a0 = *(const float4*)(wp + 8 * j);
    float4 a1 = *(const float4*)(wp + 8 * j + 4);
    Ws[row][c16 + 8 * j + 0] = f2bf(a0.x); Ws[row][c16 + 8 * j + 1] = f2bf(a0.y);
    Ws[row][c16 + 8 * j + 2] = f2bf(a0.z); Ws[row][c16 + 8 * j + 3] = f2bf(a0.w);
    Ws[row][c16 + 8 * j + 4] = f2bf(a1.x); Ws[row][c16 + 8 * j + 5] = f2bf(a1.y);
    Ws[row][c16 + 8 * j + 6] = f2bf(a1.z); Ws[row][c16 + 8 * j + 7] = f2bf(a1.w);
  }
  __syncthreads();
  #pragma unroll
  for (int j = 0; j < 2; ++j) {
    const int chunk = tid + 256 * j;        // 0..511
    const int n = chunk >> 3;               // 0..63
    const int kk = (chunk & 7) * 8;
    union { unsigned short s[8]; uint4 u; } pk;
    #pragma unroll
    for (int i = 0; i < 8; ++i) pk.s[i] = Ws[kk + i][n];
    *(uint4*)&Wt[(size_t)(nbase + n) * C_ + kbase + kk] = pk.u;
  }
}

// ---------------------------------------------------------------------------
// Tiled WMMA GEMM: A[M][1024] bf16 x Bt[N][1024] bf16 (pre-transposed) + bias.
// Block 256 thr (8 waves), tile 128x64, K-stage 64, double-buffered async LDS.
// EPI 0: scatter to Q/K [B,H,T,D], V^T [B,H,D,T] bf16.  EPI 1: f32 out.
// ---------------------------------------------------------------------------
template <int EPI>
__global__ __launch_bounds__(256) void gemm_kernel(
    const unsigned short* __restrict__ A, const unsigned short* __restrict__ Bt,
    const float* __restrict__ bias,
    unsigned short* __restrict__ Qo, unsigned short* __restrict__ Ko,
    unsigned short* __restrict__ VTo, float* __restrict__ Out)
{
  __shared__ unsigned short As[2][128][72];   // 36 KB
  __shared__ unsigned short Bs[2][64][72];    // 18 KB

  const int tid  = threadIdx.x;
  const int lane = tid & 31;
  const int wave = tid >> 5;                  // M sub-tile 0..7
  const int ln = lane & 15, g = lane >> 4;
  const int nbase = blockIdx.x * 64;
  const int mbase = blockIdx.y * 128;

  v8f acc[4] = {};

  auto issueStage = [&](int buf, int k0) {    // 6 async b128 copies / thread
    #pragma unroll
    for (int j = 0; j < 4; ++j) {
      const int chunk = tid + 256 * j;        // 0..1023
      const int row = chunk >> 3, c8 = (chunk & 7) * 8;
      cp16(&A[(size_t)(mbase + row) * C_ + k0 + c8], &As[buf][row][c8]);
    }
    #pragma unroll
    for (int j = 0; j < 2; ++j) {
      const int chunk = tid + 256 * j;        // 0..511
      const int row = chunk >> 3, c8 = (chunk & 7) * 8;
      cp16(&Bt[(size_t)(nbase + row) * C_ + k0 + c8], &Bs[buf][row][c8]);
    }
  };

  auto compute = [&](int buf) {               // 8 WMMAs / wave / stage
    #pragma unroll
    for (int ks = 0; ks < 2; ++ks) {
      Frag16 af;
      af.u[0] = *(const uint4*)&As[buf][wave * 16 + ln][32 * ks + 8 * g];
      af.u[1] = *(const uint4*)&As[buf][wave * 16 + ln][32 * ks + 16 + 8 * g];
      #pragma unroll
      for (int t = 0; t < 4; ++t) {
        Frag16 bf;
        bf.u[0] = *(const uint4*)&Bs[buf][t * 16 + ln][32 * ks + 8 * g];
        bf.u[1] = *(const uint4*)&Bs[buf][t * 16 + ln][32 * ks + 16 + 8 * g];
        acc[t] = wmma_bf16(af, bf, acc[t]);
      }
    }
  };

  issueStage(0, 0);
  const int NST = C_ / 64;                    // 16 stages
  for (int s = 0; s < NST; ++s) {
    const int buf = s & 1;
    if (s + 1 < NST) { issueStage(buf ^ 1, (s + 1) * 64); wait_async_hi(); }
    else             { wait_async_0(); }
    __syncthreads();                          // stage s resident for all waves
    compute(buf);
    __syncthreads();                          // done reading before next fill
  }

  if (EPI == 0) {
    #pragma unroll
    for (int t = 0; t < 4; ++t) {
      const int ng = nbase + t * 16 + ln;
      const float bv = bias[ng];
      const int which = ng >> 10;             // 0=Q 1=K 2=V
      const int cc = ng & 1023;
      const int h = cc >> 6, d = cc & 63;
      #pragma unroll
      for (int r = 0; r < 8; ++r) {
        const int mg = mbase + wave * 16 + r + 8 * g;
        const int bi = mg / T_;
        const int tq = mg - bi * T_;
        const size_t bh = (size_t)(bi * H_ + h);
        const unsigned short val = f2bf(acc[t][r] + bv);
        if (which == 0)      Qo [bh * (T_ * D_) + (size_t)tq * D_ + d] = val;
        else if (which == 1) Ko [bh * (T_ * D_) + (size_t)tq * D_ + d] = val;
        else                 VTo[bh * (D_ * T_) + (size_t)d  * T_ + tq] = val;
      }
    }
  } else {
    #pragma unroll
    for (int t = 0; t < 4; ++t) {
      const int ng = nbase + t * 16 + ln;
      const float bv = bias[ng];
      #pragma unroll
      for (int r = 0; r < 8; ++r) {
        const int mg = mbase + wave * 16 + r + 8 * g;
        Out[(size_t)mg * C_ + ng] = acc[t][r] + bv;
      }
    }
  }
}

// ---------------------------------------------------------------------------
// Causal flash attention: one wave per 16 query rows per (b,h).
// ---------------------------------------------------------------------------
__global__ __launch_bounds__(32) void attn_kernel(
    const unsigned short* __restrict__ Q, const unsigned short* __restrict__ K,
    const unsigned short* __restrict__ VT, unsigned short* __restrict__ Y)
{
  __shared__ unsigned short Ps[16][40];

  const int lane = threadIdx.x & 31;
  const int ln = lane & 15, g = lane >> 4;
  const int qbase = blockIdx.x * 16;
  const int h = blockIdx.y, b = blockIdx.z;
  const size_t bh = (size_t)(b * H_ + h);
  const unsigned short* Qp = Q  + bh * (size_t)(T_ * D_);
  const unsigned short* Kp = K  + bh * (size_t)(T_ * D_);
  const unsigned short* Vp = VT + bh * (size_t)(D_ * T_);

  Frag16 qf[2];
  #pragma unroll
  for (int s = 0; s < 2; ++s) {
    const unsigned short* p = &Qp[(size_t)(qbase + ln) * D_ + 32 * s + 8 * g];
    qf[s].u[0] = *(const uint4*)p;
    qf[s].u[1] = *(const uint4*)(p + 16);
  }

  v8f oacc[4] = {};
  float rowmax[8], rowsum[8];
  #pragma unroll
  for (int r = 0; r < 8; ++r) { rowmax[r] = -3.0e38f; rowsum[r] = 0.0f; }

  const int nkt = (qbase + 15) / 32 + 1;      // kbase <= qbase (causal)
  for (int kt = 0; kt < nkt; ++kt) {
    const int kbase = kt * 32;

    v8f s0 = {}, s1 = {};
    #pragma unroll
    for (int s = 0; s < 2; ++s) {
      Frag16 kb0, kb1;
      const unsigned short* p0 = &Kp[(size_t)(kbase + ln) * D_ + 32 * s + 8 * g];
      kb0.u[0] = *(const uint4*)p0; kb0.u[1] = *(const uint4*)(p0 + 16);
      const unsigned short* p1 = &Kp[(size_t)(kbase + 16 + ln) * D_ + 32 * s + 8 * g];
      kb1.u[0] = *(const uint4*)p1; kb1.u[1] = *(const uint4*)(p1 + 16);
      s0 = wmma_bf16(qf[s], kb0, s0);
      s1 = wmma_bf16(qf[s], kb1, s1);
    }

    float x0[8], x1[8], alpha[8];
    #pragma unroll
    for (int r = 0; r < 8; ++r) {
      const int row = qbase + r + 8 * g;
      x0[r] = (kbase + ln      <= row) ? s0[r] * (1.0f / 64.0f) : -3.0e38f;
      x1[r] = (kbase + 16 + ln <= row) ? s1[r] * (1.0f / 64.0f) : -3.0e38f;
    }
    #pragma unroll
    for (int r = 0; r < 8; ++r) {
      float t = fmaxf(x0[r], x1[r]);
      t = fmaxf(t, __shfl_xor(t, 1, 32));
      t = fmaxf(t, __shfl_xor(t, 2, 32));
      t = fmaxf(t, __shfl_xor(t, 4, 32));
      t = fmaxf(t, __shfl_xor(t, 8, 32));
      const float mnew = fmaxf(rowmax[r], t);
      alpha[r] = __expf(rowmax[r] - mnew);
      rowmax[r] = mnew;
      const float p0 = __expf(x0[r] - mnew);
      const float p1 = __expf(x1[r] - mnew);
      float ls = p0 + p1;
      ls += __shfl_xor(ls, 1, 32);
      ls += __shfl_xor(ls, 2, 32);
      ls += __shfl_xor(ls, 4, 32);
      ls += __shfl_xor(ls, 8, 32);
      rowsum[r] = rowsum[r] * alpha[r] + ls;
      Ps[r + 8 * g][ln]      = f2bf(p0);
      Ps[r + 8 * g][16 + ln] = f2bf(p1);
    }
    #pragma unroll
    for (int r = 0; r < 8; ++r) {
      oacc[0][r] *= alpha[r]; oacc[1][r] *= alpha[r];
      oacc[2][r] *= alpha[r]; oacc[3][r] *= alpha[r];
    }
    __syncthreads();

    Frag16 pf;
    pf.u[0] = *(const uint4*)&Ps[ln][8 * g];
    pf.u[1] = *(const uint4*)&Ps[ln][16 + 8 * g];
    #pragma unroll
    for (int t = 0; t < 4; ++t) {
      Frag16 vf;
      const unsigned short* vp = &Vp[(size_t)(t * 16 + ln) * T_ + kbase + 8 * g];
      vf.u[0] = *(const uint4*)vp;
      vf.u[1] = *(const uint4*)(vp + 16);
      oacc[t] = wmma_bf16(pf, vf, oacc[t]);
    }
    __syncthreads();
  }

  #pragma unroll
  for (int r = 0; r < 8; ++r) {
    const int row = qbase + r + 8 * g;
    const size_t base = ((size_t)b * T_ + row) * C_ + h * D_;
    const float inv = 1.0f / rowsum[r];
    Y[base +  0 + ln] = f2bf(oacc[0][r] * inv);
    Y[base + 16 + ln] = f2bf(oacc[1][r] * inv);
    Y[base + 32 + ln] = f2bf(oacc[2][r] * inv);
    Y[base + 48 + ln] = f2bf(oacc[3][r] * inv);
  }
}

// ---------------------------------------------------------------------------
extern "C" void kernel_launch(void* const* d_in, const int* in_sizes, int n_in,
                              void* d_out, int out_size, void* d_ws, size_t ws_size,
                              hipStream_t stream) {
  const float* emb   = (const float*)d_in[0];
  const float* wqkv  = (const float*)d_in[1];
  const float* bqkv  = (const float*)d_in[2];
  const float* wproj = (const float*)d_in[3];
  const float* bproj = (const float*)d_in[4];
  float* out = (float*)d_out;

  const size_t nQ = (size_t)B_ * H_ * T_ * D_;   // 18,874,368 (= B*T*C)
  unsigned short* q   = (unsigned short*)d_ws;
  unsigned short* k   = q   + nQ;
  unsigned short* vt  = k   + nQ;
  unsigned short* y   = vt  + nQ;
  unsigned short* xb  = y   + nQ;
  unsigned short* wqt = xb  + nQ;
  unsigned short* wpt = wqt + (size_t)N3_ * C_;  // total ~197 MB

  convert_bf16_kernel<<<(int)(nQ / 8 / 256), 256, 0, stream>>>(emb, xb, (int)(nQ / 8));
  dim3 gt1(N3_ / 64, C_ / 64);
  convert_transpose_kernel<<<gt1, 256, 0, stream>>>(wqkv, wqt, N3_);
  dim3 gt2(C_ / 64, C_ / 64);
  convert_transpose_kernel<<<gt2, 256, 0, stream>>>(wproj, wpt, C_);

  dim3 g1(N3_ / 64, (B_ * T_) / 128);            // 48 x 144
  gemm_kernel<0><<<g1, 256, 0, stream>>>(xb, wqt, bqkv, q, k, vt, nullptr);

  dim3 g2(T_ / 16, H_, B_);                      // 36 x 16 x 32
  attn_kernel<<<g2, 32, 0, stream>>>(q, k, vt, y);

  dim3 g3(C_ / 64, (B_ * T_) / 128);             // 16 x 144
  gemm_kernel<1><<<g3, 256, 0, stream>>>(y, wpt, bproj, nullptr, nullptr, nullptr, out);
}
